// SelfAttention_28638841930338
// MI455X (gfx1250) — matrix operand order repaired and checked by reading, not compiled, not run on hardware
//
#include <hip/hip_runtime.h>
#include <hip/hip_bf16.h>
#include <cstdint>

typedef __attribute__((ext_vector_type(16))) _Float16 v16h;
typedef __attribute__((ext_vector_type(4)))  _Float16 v4h;
typedef __attribute__((ext_vector_type(8)))  float    v8f;

constexpr int BATCH = 2;
constexpr int SEQ   = 2048;
constexpr int HEADS = 16;
constexpr int HDIM  = 64;
constexpr int EMB   = HEADS * HDIM;            // 1024
constexpr int NROWS = BATCH * SEQ * HEADS;     // 65536 rows of [64]

// WMMA wrapper: D = A(16x32 f16) * B(32x16 f16) + C(16x16 f32)
__device__ __forceinline__ v8f wmma_f16(v16h a, v16h b, v8f c) {
    return __builtin_amdgcn_wmma_f32_16x16x32_f16(false, a, false, b, (short)0, c,
                                                  false, false);
}

// Async 16-byte global -> LDS copy (CDNA5, tracked by ASYNCcnt).
// LDS operand is the byte offset within the wave's LDS window = addr[31:0].
__device__ __forceinline__ void async_cp16(const void* gaddr, const void* lds) {
    const uint32_t loff = (uint32_t)(uintptr_t)lds;
    asm volatile("global_load_async_to_lds_b128 %0, %1, off"
                 :: "v"(loff), "v"(gaddr) : "memory");
}

// ---------------------------------------------------------------------------
// Per-head input projection: Y[r, e] = sum_d X[r, d] * W[e, d]
// X: [NROWS, 64] f32 (contiguous view of [B, SEQ, H*D]); W: [64, 64] f32.
// Block = 128 threads = 4 waves; each wave computes a 16x64 output tile.
// ---------------------------------------------------------------------------
__global__ __launch_bounds__(128) void proj_kernel(const float* __restrict__ X,
                                                   const float* __restrict__ W,
                                                   _Float16* __restrict__ Y) {
    const int lane = threadIdx.x & 31;
    const int wv   = threadIdx.x >> 5;                 // 0..3
    const int row0 = (blockIdx.x * 4 + wv) * 16;       // 16-row tile
    const int m    = lane & 15;
    const int kh   = lane >> 4;                        // K-half select

    v8f acc[4] = {};                                   // 4 N-tiles of 16

#pragma unroll
    for (int ks = 0; ks < 2; ks++) {                   // K = 64 = 2 x 32
        v16h a;
        {
            const float* ap = X + (size_t)(row0 + m) * HDIM + ks * 32 + kh * 8;
#pragma unroll
            for (int i = 0; i < 8; i++) {
                a[i]     = (_Float16)ap[i];
                a[i + 8] = (_Float16)ap[16 + i];
            }
        }
#pragma unroll
        for (int nt = 0; nt < 4; nt++) {
            const float* wp = W + (size_t)(nt * 16 + m) * HDIM + ks * 32 + kh * 8;
            v16h b;
#pragma unroll
            for (int i = 0; i < 16; i++) b[i] = (_Float16)wp[i + (i & 8)];
            acc[nt] = wmma_f16(a, b, acc[nt]);
        }
    }

#pragma unroll
    for (int nt = 0; nt < 4; nt++)
#pragma unroll
        for (int v = 0; v < 8; v++)
            Y[(size_t)(row0 + v + 8 * kh) * HDIM + nt * 16 + m] =
                (_Float16)acc[nt][v];
}

// ---------------------------------------------------------------------------
// f32 -> f16 bulk convert (used once for Wo). 4 elements / thread.
// ---------------------------------------------------------------------------
__global__ __launch_bounds__(256) void cvt_kernel(const float* __restrict__ src,
                                                  _Float16* __restrict__ dst) {
    const int i = (blockIdx.x * 256 + threadIdx.x) * 4;
    const float4 v = *reinterpret_cast<const float4*>(src + i);
    v4h h;
    h[0] = (_Float16)v.x; h[1] = (_Float16)v.y;
    h[2] = (_Float16)v.z; h[3] = (_Float16)v.w;
    *reinterpret_cast<v4h*>(dst + i) = h;
}

// ---------------------------------------------------------------------------
// Flash attention: one wave owns a 16-row query tile; block (8 waves) streams
// double-buffered 64-key K/V tiles through LDS via async global->LDS copies.
// Online softmax, f32 accumulators. Grid: (SEQ/128, HEADS, BATCH), block 256.
// ---------------------------------------------------------------------------
constexpr int KPAD  = 72;   // padded row stride (halves): rotates banks, keeps 16B align
constexpr int NTILE = SEQ / 64;  // 32 key tiles

__global__ __launch_bounds__(256) void attn_kernel(const _Float16* __restrict__ Qp,
                                                   const _Float16* __restrict__ Kp,
                                                   const _Float16* __restrict__ Vp,
                                                   const int* __restrict__ mask,
                                                   _Float16* __restrict__ O) {
    __shared__ _Float16 sK[2][64 * KPAD];   // [buf][key][d]
    __shared__ _Float16 sV[2][64 * KPAD];   // [buf][key][d]
    __shared__ _Float16 sP[8 * 16 * KPAD];  // per-wave P staging (C->A relayout)
    __shared__ int      sMask[SEQ];

    const int tid  = threadIdx.x;
    const int lane = tid & 31;
    const int wv   = tid >> 5;              // 0..7
    const int b    = blockIdx.z;
    const int h    = blockIdx.y;
    const int q0   = blockIdx.x * 128 + wv * 16;
    const int m    = lane & 15;
    const int kh   = lane >> 4;

    // Issue a 64x64 f16 K/V tile as per-thread async b128 copies (4 ops/thread).
    auto issue_tile = [&](int kt, int buf) {
#pragma unroll
        for (int it = 0; it < 2; it++) {
            const int c   = tid + 256 * it;   // 0..511
            const int r   = c >> 3;
            const int col = (c & 7) * 8;
            const size_t g =
                (size_t)((b * SEQ + kt + r) * HEADS + h) * HDIM + col;
            async_cp16(&Kp[g], &sK[buf][r * KPAD + col]);
            async_cp16(&Vp[g], &sV[buf][r * KPAD + col]);
        }
    };

    issue_tile(0, 0);  // prefetch tile 0

    for (int i = tid; i < SEQ; i += 256) sMask[i] = mask[b * SEQ + i];

    // Q fragments, held for the whole pass
    v16h qa[2];
    {
        const _Float16* qr =
            Qp + (size_t)((b * SEQ + q0 + m) * HEADS + h) * HDIM;
#pragma unroll
        for (int ks = 0; ks < 2; ks++) {
            const _Float16* p = qr + ks * 32 + kh * 8;
#pragma unroll
            for (int i = 0; i < 8; i++) {
                qa[ks][i]     = p[i];
                qa[ks][i + 8] = p[16 + i];
            }
        }
    }

    float mrow[8], lrow[8];
    v8f   oacc[4] = {};
#pragma unroll
    for (int v = 0; v < 8; v++) { mrow[v] = -3.0e38f; lrow[v] = 0.0f; }

    for (int t = 0; t < NTILE; t++) {
        const int cur = t & 1;
        const int kt  = t * 64;
        if (t + 1 < NTILE) {
            issue_tile(kt + 64, 1 - cur);               // overlap next tile
            asm volatile("s_wait_asynccnt 4" ::: "memory");  // tile t landed
        } else {
            asm volatile("s_wait_asynccnt 0" ::: "memory");
        }
        __syncthreads();   // tile t visible to all waves (and sMask on t==0)

        const _Float16* tK = sK[cur];
        const _Float16* tV = sV[cur];

        // S = Q * K^T (16 x 64), scale + mask
        float sv[4][8];
#pragma unroll
        for (int nt = 0; nt < 4; nt++) {
            v8f s = {};
#pragma unroll
            for (int ks = 0; ks < 2; ks++) {
                const _Float16* kp = &tK[(nt * 16 + m) * KPAD + ks * 32 + kh * 8];
                v16h bf;
#pragma unroll
                for (int i = 0; i < 16; i++) bf[i] = kp[i + (i & 8)];
                s = wmma_f16(qa[ks], bf, s);
            }
            const int mk = sMask[kt + nt * 16 + m];
#pragma unroll
            for (int v = 0; v < 8; v++)
                sv[nt][v] = mk ? s[v] * 0.125f : -1.25e18f;  // (-1e19)/sqrt(64)
        }

        // Online softmax per row (row lives in one 16-lane half-wave)
#pragma unroll
        for (int v = 0; v < 8; v++) {
            float rmax = fmaxf(fmaxf(sv[0][v], sv[1][v]),
                               fmaxf(sv[2][v], sv[3][v]));
#pragma unroll
            for (int off = 1; off < 16; off <<= 1)
                rmax = fmaxf(rmax, __shfl_xor(rmax, off, 32));
            const float mnew  = fmaxf(mrow[v], rmax);
            const float alpha = __expf(mrow[v] - mnew);
            float psum = 0.0f;
#pragma unroll
            for (int nt = 0; nt < 4; nt++) {
                const float p = __expf(sv[nt][v] - mnew);
                sv[nt][v] = p;
                psum += p;
            }
#pragma unroll
            for (int off = 1; off < 16; off <<= 1)
                psum += __shfl_xor(psum, off, 32);
            lrow[v] = lrow[v] * alpha + psum;
            mrow[v] = mnew;
#pragma unroll
            for (int dt = 0; dt < 4; dt++) oacc[dt][v] *= alpha;
        }

        // Re-layout P: C-fragment -> LDS -> A-fragment (per-wave region)
        _Float16* pw = &sP[wv * 16 * KPAD];
#pragma unroll
        for (int nt = 0; nt < 4; nt++)
#pragma unroll
            for (int v = 0; v < 8; v++)
                pw[(v + 8 * kh) * KPAD + nt * 16 + m] = (_Float16)sv[nt][v];
        asm volatile("s_wait_dscnt 0" ::: "memory");

        v16h pa[2];
#pragma unroll
        for (int ks = 0; ks < 2; ks++) {
            const _Float16* p = &pw[m * KPAD + ks * 32 + kh * 8];
#pragma unroll
            for (int i = 0; i < 8; i++) {
                pa[ks][i]     = p[i];
                pa[ks][i + 8] = p[16 + i];
            }
        }

        // O += P * V  (B fragment: fixed column d = dt*16+m, strided over keys)
#pragma unroll
        for (int dt = 0; dt < 4; dt++)
#pragma unroll
            for (int ks = 0; ks < 2; ks++) {
                v16h vb;
#pragma unroll
                for (int i = 0; i < 16; i++)
                    vb[i] = tV[(ks * 32 + kh * 8 + i + (i & 8)) * KPAD + dt * 16 + m];
                oacc[dt] = wmma_f16(pa[ks], vb, oacc[dt]);
            }

        __syncthreads();   // tile t fully consumed before buf reuse at t+1
    }

    // Normalize and store merged heads: [B, SEQ, EMB] f16
#pragma unroll
    for (int v = 0; v < 8; v++) {
        const float inv = 1.0f / lrow[v];
        const int   r   = q0 + v + 8 * kh;
#pragma unroll
        for (int dt = 0; dt < 4; dt++)
            O[(size_t)(b * SEQ + r) * EMB + h * HDIM + dt * 16 + m] =
                (_Float16)(oacc[dt][v] * inv);
    }
}

// ---------------------------------------------------------------------------
// Output projection: Y[r, e] = sum_c X[r, c] * Wo16[e, c] + bo[e]
// Block = 256 (8 waves): 128 rows x 64 cols; waves share the Wo column band.
// Grid: (BATCH*SEQ/128, EMB/64)
// ---------------------------------------------------------------------------
__global__ __launch_bounds__(256) void outproj_kernel(const _Float16* __restrict__ X,
                                                      const _Float16* __restrict__ Wo16,
                                                      const float* __restrict__ bo,
                                                      float* __restrict__ Y) {
    const int lane = threadIdx.x & 31;
    const int wv   = threadIdx.x >> 5;
    const int row0 = blockIdx.x * 128 + wv * 16;
    const int e0   = blockIdx.y * 64;
    const int m    = lane & 15;
    const int kh   = lane >> 4;

    v8f acc[4] = {};
    for (int ks = 0; ks < EMB / 32; ks++) {
        v16h a;
        const _Float16* xp = X + (size_t)(row0 + m) * EMB + ks * 32 + kh * 8;
#pragma unroll
        for (int i = 0; i < 8; i++) {
            a[i]     = xp[i];
            a[i + 8] = xp[16 + i];
        }
#pragma unroll
        for (int nt = 0; nt < 4; nt++) {
            const _Float16* wp =
                Wo16 + (size_t)(e0 + nt * 16 + m) * EMB + ks * 32 + kh * 8;
            v16h bf;
#pragma unroll
            for (int i = 0; i < 16; i++) bf[i] = wp[i + (i & 8)];
            acc[nt] = wmma_f16(a, bf, acc[nt]);
        }
    }

#pragma unroll
    for (int nt = 0; nt < 4; nt++) {
        const int   e    = e0 + nt * 16 + m;
        const float bias = bo[e];
#pragma unroll
        for (int v = 0; v < 8; v++)
            Y[(size_t)(row0 + v + 8 * kh) * EMB + e] = acc[nt][v] + bias;
    }
}

// ---------------------------------------------------------------------------
extern "C" void kernel_launch(void* const* d_in, const int* in_sizes, int n_in,
                              void* d_out, int out_size, void* d_ws, size_t ws_size,
                              hipStream_t stream) {
    const float* values  = (const float*)d_in[0];
    const float* keys    = (const float*)d_in[1];
    const float* queries = (const float*)d_in[2];
    const int*   mask    = (const int*)d_in[3];
    const float* Wv      = (const float*)d_in[4];
    const float* Wk      = (const float*)d_in[5];
    const float* Wq      = (const float*)d_in[6];
    const float* Wo      = (const float*)d_in[7];
    const float* bo      = (const float*)d_in[8];
    float*       out     = (float*)d_out;

    // Workspace: qp | kp | vp (8MB each, f16) | attn_out (8MB) | Wo f16 (2MB)
    _Float16* qp  = (_Float16*)d_ws;
    _Float16* kp  = qp  + (size_t)NROWS * HDIM;
    _Float16* vp  = kp  + (size_t)NROWS * HDIM;
    _Float16* ao  = vp  + (size_t)NROWS * HDIM;
    _Float16* wo16 = ao + (size_t)BATCH * SEQ * EMB;

    const dim3 pblk(128);
    const dim3 pgrd(NROWS / 64);  // 4 waves x 16 rows per block
    proj_kernel<<<pgrd, pblk, 0, stream>>>(queries, Wq, qp);
    proj_kernel<<<pgrd, pblk, 0, stream>>>(keys,    Wk, kp);
    proj_kernel<<<pgrd, pblk, 0, stream>>>(values,  Wv, vp);

    cvt_kernel<<<dim3((EMB * EMB) / 1024), dim3(256), 0, stream>>>(Wo, wo16);

    attn_kernel<<<dim3(SEQ / 128, HEADS, BATCH), dim3(256), 0, stream>>>(
        qp, kp, vp, mask, ao);

    outproj_kernel<<<dim3(BATCH * SEQ / 128, EMB / 64), dim3(256), 0, stream>>>(
        ao, wo16, bo, out);
}
